// GRU_Attention_Sentence_26414048870690
// MI455X (gfx1250) — matrix operand order repaired and checked by reading, not compiled, not run on hardware
//
#include <hip/hip_runtime.h>
#include <cstdint>
#include <cstddef>

// ---------------------------------------------------------------------------
// Problem constants (from reference)
// ---------------------------------------------------------------------------
constexpr int Bb = 128;        // batch
constexpr int Ss = 256;        // seq len
constexpr int Ee = 300;        // embedding dim
constexpr int Hh = 512;        // hidden
constexpr int Dd = 16;         // domains
constexpr int EP = 320;        // E padded to multiple of 32 (WMMA K)
constexpr int G3 = 3 * Hh;     // 1536 gate width
constexpr int ROWS = Ss * Bb;  // 32768 time-major rows

// ---------------------------------------------------------------------------
// Workspace layout (bytes; every size is a multiple of 256)
// ---------------------------------------------------------------------------
constexpr size_t OFF_XSA  = 0;                                   // bf16 [ROWS][EP]
constexpr size_t SZ_XSA   = (size_t)ROWS * EP * 2;
constexpr size_t OFF_WIH  = OFF_XSA + SZ_XSA;                    // bf16 [2][G3][EP]
constexpr size_t SZ_WIH   = (size_t)2 * G3 * EP * 2;
constexpr size_t OFF_WHH  = OFF_WIH + SZ_WIH;                    // bf16 [2][G3][H]
constexpr size_t SZ_WHH   = (size_t)2 * G3 * Hh * 2;
constexpr size_t OFF_GI   = OFF_WHH + SZ_WHH;                    // f32 [2][ROWS][G3]
constexpr size_t SZ_GI    = (size_t)2 * ROWS * G3 * 4;
constexpr size_t OFF_HBF  = OFF_GI + SZ_GI;                      // bf16 [2dir][2buf][B][H]
constexpr size_t SZ_HBF   = (size_t)4 * Bb * Hh * 2;
constexpr size_t OFF_HF   = OFF_HBF + SZ_HBF;                    // f32  [2dir][2buf][B][H]
constexpr size_t SZ_HF    = (size_t)4 * Bb * Hh * 4;
constexpr size_t OFF_HCAT = OFF_HF + SZ_HF;                      // f32 [B][S][2H]
constexpr size_t SZ_HCAT  = (size_t)Bb * Ss * 2 * Hh * 4;
constexpr size_t OFF_CTR  = OFF_HCAT + SZ_HCAT;                  // int[32] spin-barrier ctrs

// ---------------------------------------------------------------------------
// WMMA types/helpers (CDNA5: V_WMMA_F32_16X16X32_BF16, wave32)
// ---------------------------------------------------------------------------
typedef __bf16 v16bf __attribute__((ext_vector_type(16)));
typedef __bf16 v8bf  __attribute__((ext_vector_type(8)));
typedef float  v8f   __attribute__((ext_vector_type(8)));
typedef int    v4i   __attribute__((vector_size(16)));   // async-LDS builtin pointee

union AFrag {
    v16bf v;
    struct { v8bf lo; v8bf hi; } p;
};

__device__ inline v8f wmma_bf16(v16bf a, v16bf b, v8f c) {
    // 8 args: (neg_a, A, neg_b, B, c_mod, C, reuse_a, reuse_b)
    return __builtin_amdgcn_wmma_f32_16x16x32_bf16(false, a, false, b,
                                                   (short)0, c, false, false);
}

__device__ inline float sigmoidf_(float x) { return 1.0f / (1.0f + __expf(-x)); }

// ---------------------------------------------------------------------------
// Async global->LDS copy (GLOBAL_LOAD_ASYNC_TO_LDS_B128, ASYNCcnt-tracked).
// Guarded: falls back to load+ds_store if the toolchain lacks the builtins.
// ---------------------------------------------------------------------------
#if defined(__has_builtin)
#  if __has_builtin(__builtin_amdgcn_global_load_async_to_lds_b128) && \
      __has_builtin(__builtin_amdgcn_s_wait_asynccnt)
#    define HAVE_ASYNC_LDS 1
#  endif
#endif
#ifndef HAVE_ASYNC_LDS
#  define HAVE_ASYNC_LDS 0
#endif

__device__ inline void lds_copy16_async(const __bf16* g, __bf16* l) {
#if HAVE_ASYNC_LDS
    // Builtin expects pointers to 16-byte int vectors (per compiler diagnostic).
    __builtin_amdgcn_global_load_async_to_lds_b128((v4i*)g, (v4i*)l, 0, 0);
#else
    *(v8bf*)l = *(const v8bf*)g;
#endif
}

__device__ inline void lds_copy_join() {
#if HAVE_ASYNC_LDS
    __builtin_amdgcn_s_wait_asynccnt(0);
#endif
}

// ---------------------------------------------------------------------------
// k_init: zero h state buffers + barrier counters (runs every graph replay)
// ---------------------------------------------------------------------------
__global__ void k_init(uint32_t* hstate_words, size_t nwords, int* ctr) {
    size_t stride = (size_t)gridDim.x * blockDim.x;
    for (size_t i = (size_t)blockIdx.x * blockDim.x + threadIdx.x; i < nwords; i += stride)
        hstate_words[i] = 0u;
    if (blockIdx.x == 0 && threadIdx.x < 32) ctr[threadIdx.x] = 0;
}

// ---------------------------------------------------------------------------
// k_prep: fp32 weights -> bf16 (W_ih padded E->EP with zeros)
// ---------------------------------------------------------------------------
__global__ void k_prep(const float* wihF, const float* wihB,
                       const float* whhF, const float* whhB,
                       __bf16* WIH, __bf16* WHH) {
    int stride = gridDim.x * blockDim.x;
    int t0 = blockIdx.x * blockDim.x + threadIdx.x;
    for (int i = t0; i < 2 * G3 * EP; i += stride) {
        int dir = i / (G3 * EP);
        int rem = i % (G3 * EP);
        int r = rem / EP, k = rem % EP;
        const float* W = dir ? wihB : wihF;
        WIH[i] = (k < Ee) ? (__bf16)W[r * Ee + k] : (__bf16)0.0f;
    }
    for (int i = t0; i < 2 * G3 * Hh; i += stride) {
        int dir = i / (G3 * Hh);
        int rem = i % (G3 * Hh);
        const float* W = dir ? whhB : whhF;
        WHH[i] = (__bf16)W[rem];
    }
}

// ---------------------------------------------------------------------------
// k_embed: gather emb[x] into time-major bf16 xs[row = s*B + b][EP]
// ---------------------------------------------------------------------------
__global__ void k_embed(const int* x, const float* emb, __bf16* xsA) {
    int row = blockIdx.x;            // 0..ROWS-1, row = s*B + b
    int s = row / Bb, b = row % Bb;
    int tok = x[b * Ss + s];
    const float* erow = emb + (size_t)tok * Ee;
    __bf16* orow = xsA + (size_t)row * EP;
    for (int k = threadIdx.x; k < EP; k += blockDim.x)
        orow[k] = (k < Ee) ? (__bf16)erow[k] : (__bf16)0.0f;
}

// ---------------------------------------------------------------------------
// k_gemm: gi[dir] = xs @ W_ih[dir]^T + b_ih   ([ROWS,EP] x [G3,EP] -> [ROWS,G3])
// 8 waves/block; wave => one 16-row M tile, 4 N tiles (16x64 strip); K loop 10.
// The 64x320 bf16 W strip (40KB) is shared by all 8 waves -> staged once in
// LDS via async global->LDS copies; B fragments then come from ds_load.
// ---------------------------------------------------------------------------
__global__ __launch_bounds__(256) void k_gemm(const __bf16* __restrict__ xsA,
                                              const __bf16* __restrict__ WIH,
                                              const float* bihF, const float* bihB,
                                              float* giF, float* giB) {
    __shared__ __bf16 sB[64 * EP];                 // 40 KB

    const int dir = blockIdx.z;
    const __bf16* W = WIH + (size_t)dir * G3 * EP;
    const float* bih = dir ? bihB : bihF;
    float* gi = dir ? giB : giF;

    const int n0 = blockIdx.x * 64;                // output col strip base

    // Stage W rows [n0, n0+64) -> LDS (contiguous 40KB block), 16B per lane.
    {
        const __bf16* Wstrip = W + (size_t)n0 * EP;
        for (int i = threadIdx.x; i < (64 * EP) / 8; i += 256)
            lds_copy16_async(Wstrip + i * 8, sB + i * 8);
        lds_copy_join();
    }
    __syncthreads();

    const int wv = threadIdx.x >> 5;
    const int ln = threadIdx.x & 31;
    const int lm = ln & 15;          // lane's row/col within tile
    const int kh = ln >> 4;          // K-half select (A/B fragment striping)

    const int m0 = (blockIdx.y * 8 + wv) * 16;     // output row tile base

    v8f acc[4] = {{}, {}, {}, {}};
    const __bf16* arow = xsA + (size_t)(m0 + lm) * EP + (kh << 4);

    for (int k0 = 0; k0 < EP; k0 += 32) {
        __builtin_prefetch(arow + k0 + 32, 0, 1);
        AFrag a;
        a.p.lo = *(const v8bf*)(arow + k0);        // K = k0 + {0..7}  (+16 for hi lanes)
        a.p.hi = *(const v8bf*)(arow + k0 + 8);    // K = k0 + {8..15} (+16 for hi lanes)
#pragma unroll
        for (int i = 0; i < 4; ++i) {
            const v16bf* bp =
                (const v16bf*)(sB + (size_t)(i * 16 + lm) * EP + k0 + (kh << 4));
            acc[i] = wmma_bf16(a.v, *bp, acc[i]);
        }
    }

#pragma unroll
    for (int i = 0; i < 4; ++i) {
        int n = n0 + i * 16 + lm;
        float bias = bih[n];
        int rbase = m0 + (kh << 3);   // C/D layout: VGPR g -> row g (+8 for hi lanes)
#pragma unroll
        for (int g = 0; g < 8; ++g)
            gi[(size_t)(rbase + g) * G3 + n] = acc[i][g] + bias;
    }
}

// ---------------------------------------------------------------------------
// k_recur: sequential GRU scan, 8 WGs per direction (grid {8,2}), 1024 thr/WG.
// Each WG owns 64 hidden units; stages its 192x512 W_hh slice (196KB) in LDS
// via async global->LDS copies, reused for all 256 steps.
// h double-buffered in global; device-scope atomic split-barrier between steps.
// Per wave per step: 16 K-steps x 3 gates = 48 WMMAs.
// ---------------------------------------------------------------------------
__global__ __launch_bounds__(1024) void k_recur(const __bf16* __restrict__ WHH,
                                                const float* bhhF, const float* bhhB,
                                                const float* __restrict__ giF,
                                                const float* __restrict__ giB,
                                                __bf16* hbf_base, float* hf_base,
                                                float* __restrict__ hcat, int* ctr) {
    extern __shared__ char smem_raw[];
    __bf16* sW = (__bf16*)smem_raw;                 // [192][512] bf16 = 196608 B

    const int wg  = blockIdx.x;                     // 0..7 : hidden-unit slice
    const int dir = blockIdx.y;                     // 0 fwd, 1 bwd
    const __bf16* Whh = WHH + (size_t)dir * G3 * Hh;
    const float* bhh = dir ? bhhB : bhhF;
    const float* gi  = dir ? giB : giF;
    __bf16* hbfs = hbf_base + (size_t)dir * 2 * Bb * Hh;
    float*  hfs  = hf_base  + (size_t)dir * 2 * Bb * Hh;
    int* c = ctr + dir * 16;                        // separate cachelines per dir

    const int jbase = wg * 64;                      // this WG's hidden range [jbase, jbase+64)

    // Stage W_hh rows {j, 512+j, 1024+j : j in slice} into LDS, [gate*64+jl][k].
    // 12288 x 16B async chunks (rows are 512 elems, so chunks never straddle rows).
    for (int i = threadIdx.x; i < (192 * 512) / 8; i += 1024) {
        int e = i * 8;
        int lr = e >> 9, k = e & 511;
        int gate = lr >> 6, jl = lr & 63;
        lds_copy16_async(Whh + (size_t)(gate * Hh + jbase + jl) * Hh + k, sW + e);
    }
    lds_copy_join();
    __syncthreads();

    const int w  = threadIdx.x >> 5;
    const int ln = threadIdx.x & 31;
    const int lm = ln & 15;
    const int kh = ln >> 4;
    const int m  = w & 7;                           // batch tile 0..7
    const int jt = w >> 3;                          // local hidden tile 0..3
    const int jl0 = jt * 16 + lm;                   // local hidden col 0..63
    const int j   = jbase + jl0;                    // absolute hidden index
    const int brow = m * 16 + lm;                   // A-fragment row (batch)

    const float bhr = bhh[j], bhz = bhh[Hh + j], bhn = bhh[2 * Hh + j];

    for (int step = 0; step < Ss; ++step) {
        const int s = dir ? (Ss - 1 - step) : step;
        const int cur = step & 1, nxt = cur ^ 1;
        const __bf16* hin  = hbfs + (size_t)cur * Bb * Hh;
        const float*  hfin = hfs  + (size_t)cur * Bb * Hh;
        __bf16* hout  = hbfs + (size_t)nxt * Bb * Hh;
        float*  hfout = hfs  + (size_t)nxt * Bb * Hh;

        v8f ar = {}, az = {}, an = {};
        const __bf16* arow = hin + (size_t)brow * Hh + (kh << 4);
#pragma unroll 4
        for (int k0 = 0; k0 < Hh; k0 += 32) {
            AFrag a;
            a.p.lo = *(const v8bf*)(arow + k0);
            a.p.hi = *(const v8bf*)(arow + k0 + 8);
            const __bf16* bbase = sW + (size_t)(jt * 16 + lm) * 512 + k0 + (kh << 4);
            v16bf br = *(const v16bf*)(bbase);                 // gate r rows
            v16bf bz = *(const v16bf*)(bbase + 64 * 512);      // gate z rows
            v16bf bn = *(const v16bf*)(bbase + 128 * 512);     // gate n rows
            ar = wmma_bf16(a.v, br, ar);
            az = wmma_bf16(a.v, bz, az);
            an = wmma_bf16(a.v, bn, an);
        }

        const size_t gibase = ((size_t)s * Bb) * G3;
#pragma unroll
        for (int g = 0; g < 8; ++g) {
            int b = m * 16 + g + (kh << 3);
            size_t gidx = gibase + (size_t)b * G3 + j;
            float ir = gi[gidx], iz = gi[gidx + Hh], in_ = gi[gidx + 2 * Hh];
            float r  = sigmoidf_(ir + ar[g] + bhr);
            float zg = sigmoidf_(iz + az[g] + bhz);
            float nn = tanhf(in_ + r * (an[g] + bhn));
            float hp = hfin[(size_t)b * Hh + j];
            float hn = (1.0f - zg) * nn + zg * hp;
            hfout[(size_t)b * Hh + j] = hn;
            hout[(size_t)b * Hh + j]  = (__bf16)hn;
            hcat[((size_t)b * Ss + s) * (2 * Hh) + dir * Hh + j] = hn;
        }

        // --- device-scope split barrier across the 8 WGs of this direction ---
        __threadfence();
        __syncthreads();
        if (threadIdx.x == 0) {
            __hip_atomic_fetch_add(c, 1, __ATOMIC_RELEASE, __HIP_MEMORY_SCOPE_AGENT);
            int target = 8 * (step + 1);
            while (__hip_atomic_load(c, __ATOMIC_ACQUIRE, __HIP_MEMORY_SCOPE_AGENT) < target)
                __builtin_amdgcn_s_sleep(2);
        }
        __syncthreads();
        __threadfence();
    }
}

// ---------------------------------------------------------------------------
// k_att: per batch row: softmax(att_w[:, z[b]]) over 2H, matvec with h, FC.
// ---------------------------------------------------------------------------
__global__ __launch_bounds__(256) void k_att(const float* __restrict__ hcat,
                                             const float* __restrict__ att_w,
                                             const int* __restrict__ z,
                                             const float* __restrict__ fcw,
                                             const float* __restrict__ fcb,
                                             float* __restrict__ out) {
    __shared__ float sa[2 * Hh];    // softmax weights
    __shared__ float red[256];
    __shared__ float satt[Ss];
    const int b = blockIdx.x;
    const int t = threadIdx.x;
    const int zb = z[b];

    for (int d = t; d < 2 * Hh; d += 256) sa[d] = att_w[d * Dd + zb];
    __syncthreads();

    float mv = -3.402823e38f;
    for (int d = t; d < 2 * Hh; d += 256) mv = fmaxf(mv, sa[d]);
    red[t] = mv; __syncthreads();
    for (int off = 128; off > 0; off >>= 1) {
        if (t < off) red[t] = fmaxf(red[t], red[t + off]);
        __syncthreads();
    }
    const float M = red[0]; __syncthreads();

    float ssum = 0.0f;
    for (int d = t; d < 2 * Hh; d += 256) {
        float e = __expf(sa[d] - M);
        sa[d] = e;
        ssum += e;
    }
    red[t] = ssum; __syncthreads();
    for (int off = 128; off > 0; off >>= 1) {
        if (t < off) red[t] += red[t + off];
        __syncthreads();
    }
    const float inv = 1.0f / red[0]; __syncthreads();

    // att[b, s=t] = <h[b,t,:], softmax>
    const float* hrow = hcat + ((size_t)b * Ss + t) * (2 * Hh);
    float acc = 0.0f;
    for (int d = 0; d < 2 * Hh; ++d) acc += hrow[d] * sa[d];
    satt[t] = acc * inv;
    __syncthreads();

    float p0 = satt[t] * fcw[t];
    float p1 = satt[t] * fcw[Ss + t];
    red[t] = p0; __syncthreads();
    for (int off = 128; off > 0; off >>= 1) {
        if (t < off) red[t] += red[t + off];
        __syncthreads();
    }
    float o0 = red[0]; __syncthreads();
    red[t] = p1; __syncthreads();
    for (int off = 128; off > 0; off >>= 1) {
        if (t < off) red[t] += red[t + off];
        __syncthreads();
    }
    if (t == 0) {
        out[b * 2 + 0] = o0 + fcb[0];
        out[b * 2 + 1] = red[0] + fcb[1];
    }
}

// ---------------------------------------------------------------------------
// kernel_launch
// ---------------------------------------------------------------------------
extern "C" void kernel_launch(void* const* d_in, const int* in_sizes, int n_in,
                              void* d_out, int out_size, void* d_ws, size_t ws_size,
                              hipStream_t stream) {
    (void)in_sizes; (void)n_in; (void)out_size; (void)ws_size;

    const int*   x     = (const int*)d_in[0];
    const int*   z     = (const int*)d_in[1];
    const float* emb   = (const float*)d_in[2];
    const float* wihF  = (const float*)d_in[3];
    const float* whhF  = (const float*)d_in[4];
    const float* bihF  = (const float*)d_in[5];
    const float* bhhF  = (const float*)d_in[6];
    const float* wihB  = (const float*)d_in[7];
    const float* whhB  = (const float*)d_in[8];
    const float* bihB  = (const float*)d_in[9];
    const float* bhhB  = (const float*)d_in[10];
    const float* att_w = (const float*)d_in[11];
    const float* fcw   = (const float*)d_in[12];
    const float* fcb   = (const float*)d_in[13];
    float* out = (float*)d_out;

    char* ws = (char*)d_ws;
    __bf16* xsA  = (__bf16*)(ws + OFF_XSA);
    __bf16* WIH  = (__bf16*)(ws + OFF_WIH);
    __bf16* WHH  = (__bf16*)(ws + OFF_WHH);
    float*  giF  = (float*)(ws + OFF_GI);
    float*  giB  = giF + (size_t)ROWS * G3;
    __bf16* hbf  = (__bf16*)(ws + OFF_HBF);
    float*  hf   = (float*)(ws + OFF_HF);
    float*  hcat = (float*)(ws + OFF_HCAT);
    int*    ctr  = (int*)(ws + OFF_CTR);

    // 1) zero h state + barrier counters (part of the captured graph)
    k_init<<<dim3(256), dim3(256), 0, stream>>>(
        (uint32_t*)(ws + OFF_HBF), (SZ_HBF + SZ_HF) / 4, ctr);

    // 2) bf16 weight conversion (pad E 300->320)
    k_prep<<<dim3(512), dim3(256), 0, stream>>>(wihF, wihB, whhF, whhB, WIH, WHH);

    // 3) embedding gather, time-major bf16
    k_embed<<<dim3(ROWS), dim3(256), 0, stream>>>(x, emb, xsA);

    // 4) input projections gi (both directions) via WMMA, B strip in LDS
    k_gemm<<<dim3(G3 / 64, ROWS / (8 * 16), 2), dim3(256), 0, stream>>>(
        xsA, WIH, bihF, bihB, giF, giB);

    // 5) sequential GRU scan, 16 WGPs busy, W_hh slice in 196KB LDS
    k_recur<<<dim3(8, 2), dim3(1024), 192 * 512 * 2, stream>>>(
        WHH, bhhF, bhhB, giF, giB, hbf, hf, hcat, ctr);

    // 6) attention + FC
    k_att<<<dim3(Bb), dim3(256), 0, stream>>>(hcat, att_w, z, fcw, fcb, out);
}
